// MultiSourceAttentionBlock_67748814127354
// MI455X (gfx1250) — compile-verified, matrix operand order checked by reading, not run on hardware
//
#include <hip/hip_runtime.h>
#include <hip/hip_bf16.h>
#include <cstdint>
#include <cstddef>

// ---------------------------------------------------------------------------
// MultiSourceAttentionBlock for MI455X (gfx1250): all large GEMMs via
// v_wmma_f32_16x16x32_bf16 with double-buffered LDS staging through
// global_load_async_to_lds_b128 (ASYNCcnt-pipelined).
// ---------------------------------------------------------------------------

typedef __attribute__((ext_vector_type(16))) __bf16 v16bf;
typedef __attribute__((ext_vector_type(8)))  __bf16 v8bf;
typedef __attribute__((ext_vector_type(8)))  float  v8f;

#define D_   1024
#define NH_  16
#define HD_  64
#define FF_  4096
#define B_   4
#define T_   512
#define TB_  1024
#define TR_  512
#define TA_  64
#define BH_  (B_*NH_)

// ---------------- helpers ----------------

__device__ __forceinline__ __bf16 f2bf(float x) {
    union { float f; uint32_t u; } cv; cv.f = x;
    uint32_t r = (cv.u + 0x7FFFu + ((cv.u >> 16) & 1u)) >> 16;
    union { unsigned short s; __bf16 b; } ov; ov.s = (unsigned short)r;
    return ov.b;
}

__device__ __forceinline__ float gelu_exact(float x) {
    return 0.5f * x * (1.0f + erff(x * 0.70710678118654752f));
}

__device__ __forceinline__ void async_copy_b128(uint32_t lds_off, const void* gptr) {
    // CDNA5 async global->LDS (tracked by ASYNCcnt)
    asm volatile("global_load_async_to_lds_b128 %0, %1, off"
                 :: "v"(lds_off), "v"(gptr) : "memory");
}

template <int N>
__device__ __forceinline__ void wait_async() {
    asm volatile("s_wait_asynccnt %0" :: "i"(N) : "memory");
}

__device__ __forceinline__ int imin(int a, int b) { return a < b ? a : b; }

// Load one 16x32 bf16 fragment (A-layout) from an LDS tile with row stride 64
// halfs. kk selects the K sub-tile (0 or 32).
__device__ __forceinline__ v16bf frag_ld64(const __bf16* tile, int rowBase, int kk, int lane) {
    int r = rowBase + (lane & 15);
    int k = kk + (lane >> 4) * 8;   // lanes 0-15: K 0..7/16..23, 16-31: 8..15/24..31
    const __bf16* p = tile + r * 64;
    v8bf lo = *(const v8bf*)(p + k);
    v8bf hi = *(const v8bf*)(p + k + 16);
    v16bf out;
#pragma unroll
    for (int i = 0; i < 8; ++i) { out[i] = lo[i]; out[i + 8] = hi[i]; }
    return out;
}

// ---------------- main WMMA GEMM ----------------
// C[b,m,n] = epi( sum_k A[b,m,k]*W[b,n,k] + bias[n] ) (+ addSrc[b,m,n])
// A: (batch,M,K) bf16 row-major lda; W: (batch,N,K) bf16 row-major ldw.
// Block tile 64x128, K-step 64, double-buffered async LDS staging;
// 8 waves (4x2), each wave computes 16x64 (8 WMMAs per K-step).
#define BM_ 64
#define BN_ 128
#define BK_ 64

template <int EPI>  // 0 none, 1 tanh, 2 gelu
__global__ __launch_bounds__(256) void k_gemm_bf16(
    const __bf16* __restrict__ A, long strideA, int lda,
    const __bf16* __restrict__ W, long strideW, int ldw,
    const float* __restrict__ bias,
    const float* __restrict__ addSrc,
    float* __restrict__ C, long strideC, int ldc,
    int M, int N, int K)
{
    __shared__ __bf16 sA[2][BM_ * BK_];   // 2 x 8 KB
    __shared__ __bf16 sB[2][BN_ * BK_];   // 2 x 16 KB

    const int bz = blockIdx.z;
    A += (long)bz * strideA;
    W += (long)bz * strideW;
    C += (long)bz * strideC;
    const float* aS = addSrc ? addSrc + (long)bz * strideC : nullptr;

    const int tid  = threadIdx.x;
    const int wave = tid >> 5;
    const int lane = tid & 31;
    const int wm   = wave >> 1;     // 0..3
    const int wn   = wave & 1;      // 0..1
    const int mBase = blockIdx.x * BM_;
    const int nBase = blockIdx.y * BN_;

    v8f acc[4];
#pragma unroll
    for (int j = 0; j < 4; ++j)
#pragma unroll
        for (int r = 0; r < 8; ++r) acc[j][r] = 0.0f;

    const uint32_t ldsA0 = (uint32_t)(uintptr_t)&sA[0][0];
    const uint32_t ldsA1 = (uint32_t)(uintptr_t)&sA[1][0];
    const uint32_t ldsB0 = (uint32_t)(uintptr_t)&sB[0][0];
    const uint32_t ldsB1 = (uint32_t)(uintptr_t)&sB[1][0];

    // Stage one 64-wide K tile into LDS buffer `buf` (6 async b128 per thread).
    auto stage = [&](int buf, int k0) {
        uint32_t la = buf ? ldsA1 : ldsA0;
        uint32_t lb = buf ? ldsB1 : ldsB0;
        // A: 64 rows x 128B -> 512 chunks of 16B, 2 per thread
#pragma unroll
        for (int i = 0; i < 2; ++i) {
            int idx = tid * 2 + i;
            int row = idx >> 3, c = idx & 7;
            int gr = imin(mBase + row, M - 1);
            async_copy_b128(la + (uint32_t)(row * 128 + c * 16),
                            A + (long)gr * lda + k0 + c * 8);
        }
        // B: 128 rows -> 1024 chunks, 4 per thread
#pragma unroll
        for (int i = 0; i < 4; ++i) {
            int idx = tid * 4 + i;
            int row = idx >> 3, c = idx & 7;
            int gr = imin(nBase + row, N - 1);
            async_copy_b128(lb + (uint32_t)(row * 128 + c * 16),
                            W + (long)gr * ldw + k0 + c * 8);
        }
    };

    const int nk = K / BK_;
    stage(0, 0);

    for (int ki = 0; ki < nk; ++ki) {
        const int buf = ki & 1;
        const bool hasNext = (ki + 1 < nk);
        if (hasNext) {
            stage(buf ^ 1, (ki + 1) * BK_);          // overlap copy of tile i+1
            if (ki + 2 < nk) {                        // L2 prefetch 2 tiles ahead
                int row = tid >> 3;
                __builtin_prefetch(A + (long)imin(mBase + row, M - 1) * lda + (ki + 2) * BK_, 0, 1);
                __builtin_prefetch(W + (long)imin(nBase + row, N - 1) * ldw + (ki + 2) * BK_, 0, 1);
            }
            wait_async<6>();   // only tile i+1's 6 loads may remain outstanding
        } else {
            wait_async<0>();
        }
        __syncthreads();       // tile i fully visible to all waves

        const __bf16* tA = &sA[buf][0];
        const __bf16* tB = &sB[buf][0];
#pragma unroll
        for (int kk = 0; kk < 2; ++kk) {
            v16bf afrag = frag_ld64(tA, wm * 16, kk * 32, lane);
#pragma unroll
            for (int j = 0; j < 4; ++j) {
                v16bf bfrag = frag_ld64(tB, wn * 64 + j * 16, kk * 32, lane);
                acc[j] = __builtin_amdgcn_wmma_f32_16x16x32_bf16(
                    false, afrag, false, bfrag, (short)0, acc[j], false, false);
            }
        }
        __syncthreads();       // all waves done reading buf before it is reused
    }

    // ---- epilogue: C layout VGPR r -> M=r (+8 for lanes 16-31), N=lane&15 ----
    const int rowSel = (lane >> 4) * 8;
    const int col    = lane & 15;
#pragma unroll
    for (int j = 0; j < 4; ++j) {
        int n = nBase + wn * 64 + j * 16 + col;
        float bv = 0.0f;
        if (bias != nullptr && n < N) bv = bias[n];
#pragma unroll
        for (int r = 0; r < 8; ++r) {
            int m = mBase + wm * 16 + rowSel + r;
            if (m < M && n < N) {
                float v = acc[j][r] + bv;
                if (EPI == 1) v = tanhf(v);
                else if (EPI == 2) v = gelu_exact(v);
                if (aS) v += aS[(long)m * ldc + n];
                C[(long)m * ldc + n] = v;
            }
        }
    }
}

// ---------------- elementwise / reduction kernels ----------------

__global__ void k_cvt_bf16(const float* __restrict__ in, __bf16* __restrict__ out, long n) {
    long i = (long)blockIdx.x * 256 + threadIdx.x;
    if (i < n) out[i] = f2bf(in[i]);
}

__global__ void k_silu(const float* __restrict__ in, float* __restrict__ out, int n) {
    int i = blockIdx.x * 256 + threadIdx.x;
    if (i < n) { float x = in[i]; out[i] = x / (1.0f + expf(-x)); }
}

// small dense: out(M,N) = act(in(M,K) @ W(N,K)^T + b); M tiny (<=4)
template <int ACT>  // 0 none, 1 gelu
__global__ void k_small_linear(const float* __restrict__ in, const float* __restrict__ W,
                               const float* __restrict__ bias, float* __restrict__ out,
                               int M, int N, int K)
{
    int n = blockIdx.x * 256 + threadIdx.x;
    if (n >= N) return;
    for (int m = 0; m < M; ++m) {
        const float* a = in + (long)m * K;
        const float* w = W + (long)n * K;
        float acc = bias ? bias[n] : 0.0f;
        for (int k = 0; k < K; ++k) acc += a[k] * w[k];
        if (ACT == 1) acc = gelu_exact(acc);
        out[(long)m * N + n] = acc;
    }
}

// LayerNorm + AdaLN modulation, writes bf16. mod row stride 6*D.
__global__ __launch_bounds__(256) void k_ln_mod(
    const float* __restrict__ x, const float* __restrict__ w, const float* __restrict__ b,
    const float* __restrict__ mod, int shiftOff, int scaleOff, int rowsPerBatch,
    __bf16* __restrict__ out)
{
    int row = blockIdx.x;
    int bidx = row / rowsPerBatch;
    const float* xr = x + (long)row * D_;
    const float* sh = mod + (long)bidx * 6 * D_ + shiftOff;
    const float* sc = mod + (long)bidx * 6 * D_ + scaleOff;
    __shared__ float r1[256], r2[256];
    int tid = threadIdx.x;
    float s = 0.f, s2 = 0.f;
    for (int i = tid; i < D_; i += 256) { float v = xr[i]; s += v; s2 += v * v; }
    r1[tid] = s; r2[tid] = s2; __syncthreads();
    for (int st = 128; st > 0; st >>= 1) {
        if (tid < st) { r1[tid] += r1[tid + st]; r2[tid] += r2[tid + st]; }
        __syncthreads();
    }
    float mean = r1[0] / D_;
    float var  = r2[0] / D_ - mean * mean;
    float inv  = rsqrtf(var + 1e-5f);
    __bf16* orow = out + (long)row * D_;
    for (int i = tid; i < D_; i += 256) {
        float v = (xr[i] - mean) * inv * w[i] + b[i];
        v = v * (1.0f + sc[i]) + sh[i];
        orow[i] = f2bf(v);
    }
}

// Per-head RMSNorm + RoPE, (B,L,D) f32 -> (B*NH, L, HD) bf16 head-major.
// One wave32 per token; each lane handles dims (lane, lane+32).
__global__ __launch_bounds__(256) void k_head_rms_rope(
    const float* __restrict__ in, const float* __restrict__ normw,
    __bf16* __restrict__ out, int L)
{
    int token = blockIdx.x * 8 + (threadIdx.x >> 5);
    int lane  = threadIdx.x & 31;
    int total = B_ * L * NH_;
    if (token >= total) return;
    int h = token % NH_;
    int l = (token / NH_) % L;
    int b = token / (NH_ * L);
    const float* src = in + ((long)(b * L + l)) * D_ + h * HD_;
    float x1 = src[lane];
    float x2 = src[lane + 32];
    float ss = x1 * x1 + x2 * x2;
#pragma unroll
    for (int m = 16; m >= 1; m >>= 1) ss += __shfl_xor(ss, m, 32);
    float inv = rsqrtf(ss * (1.0f / HD_) + 1e-6f);
    float a  = x1 * inv * normw[lane];
    float b2 = x2 * inv * normw[lane + 32];
    // RoPE: dims (lane, lane+32) share frequency lane
    float freq = (float)l * powf(10000.0f, -(float)(2 * lane) / (float)HD_);
    float cs = cosf(freq), sn = sinf(freq);
    float o1 = a * cs - b2 * sn;
    float o2 = b2 * cs + a * sn;
    __bf16* dst = out + ((long)((b * NH_ + h) * (long)L + l)) * HD_;
    dst[lane]      = f2bf(o1);
    dst[lane + 32] = f2bf(o2);
}

// V transpose: (B,L,D) f32 -> (B*NH, HD, L) bf16
__global__ void k_vt_bf16(const float* __restrict__ v, __bf16* __restrict__ vt, int L) {
    long total = (long)B_ * NH_ * HD_ * L;
    long idx = (long)blockIdx.x * 256 + threadIdx.x;
    if (idx >= total) return;
    int l = (int)(idx % L);
    int d = (int)((idx / L) % HD_);
    int h = (int)((idx / ((long)L * HD_)) % NH_);
    int b = (int)(idx / ((long)L * HD_ * NH_));
    vt[idx] = f2bf(v[((long)(b * L + l)) * D_ + h * HD_ + d]);
}

// Row softmax with temperature/bias device scalars, writes bf16 probs.
__global__ __launch_bounds__(256) void k_softmax_row(
    const float* __restrict__ s, __bf16* __restrict__ p, int L,
    const float* __restrict__ tempP, const float* __restrict__ biasP)
{
    long row = blockIdx.x;
    const float* sr = s + row * (long)L;
    __bf16* pr = p + row * (long)L;
    float temp  = 1.0f + log1pf(expf(tempP[0]));   // 1 + softplus
    float scale = temp * 0.125f;                   // temp / sqrt(HD)
    float bias  = biasP[0];
    __shared__ float red[256];
    int tid = threadIdx.x;
    float mx = -1e30f;
    for (int i = tid; i < L; i += 256) mx = fmaxf(mx, sr[i] * scale + bias);
    red[tid] = mx; __syncthreads();
    for (int st = 128; st > 0; st >>= 1) {
        if (tid < st) red[tid] = fmaxf(red[tid], red[tid + st]);
        __syncthreads();
    }
    mx = red[0]; __syncthreads();
    float sum = 0.f;
    for (int i = tid; i < L; i += 256) sum += expf(sr[i] * scale + bias - mx);
    red[tid] = sum; __syncthreads();
    for (int st = 128; st > 0; st >>= 1) {
        if (tid < st) red[tid] += red[tid + st];
        __syncthreads();
    }
    float inv = 1.0f / red[0];
    for (int i = tid; i < L; i += 256) pr[i] = f2bf(expf(sr[i] * scale + bias - mx) * inv);
}

// Combine 4 branch outputs (BH,T,HD) -> merged (B,T,D) f32
__global__ void k_combine(const float* __restrict__ os, const float* __restrict__ ob,
                          const float* __restrict__ orr, const float* __restrict__ oa,
                          const float* __restrict__ gate, const float* __restrict__ gateA,
                          float* __restrict__ merged)
{
    long idx = (long)blockIdx.x * 256 + threadIdx.x;
    if (idx >= (long)B_ * T_ * D_) return;
    int d = (int)(idx % D_);
    int t = (int)((idx / D_) % T_);
    int b = (int)(idx / ((long)D_ * T_));
    int h = d / HD_, dd = d % HD_;
    long src = ((long)(b * NH_ + h) * T_ + t) * HD_ + dd;
    float g  = tanhf(gate[0]);
    float ga = tanhf(gateA[0]);
    merged[idx] = os[src] + ob[src] + g * orr[src] + ga * oa[src];
}

__global__ void k_mean_pool(const float* __restrict__ bev, float* __restrict__ out) {
    int idx = blockIdx.x * 256 + threadIdx.x;
    if (idx >= B_ * D_) return;
    int d = idx % D_, b = idx / D_;
    float a = 0.f;
    for (int t = 0; t < TB_; ++t) a += bev[((long)(b * TB_ + t)) * D_ + d];
    out[idx] = a * (1.0f / TB_);
}

// pooled-MHA (4 heads, 1 query) helper kernels
__global__ void k_mha_scores(const float* __restrict__ q, const float* __restrict__ kk,
                             float* __restrict__ sc)
{
    int idx = blockIdx.x * 256 + threadIdx.x;   // B*4*TR
    if (idx >= B_ * 4 * TR_) return;
    int l = idx % TR_;
    int h = (idx / TR_) % 4;
    int b = idx / (TR_ * 4);
    const float* qv = q + h * 256;
    const float* kv = kk + ((long)(b * TR_ + l)) * D_ + h * 256;
    float a = 0.f;
    for (int i = 0; i < 256; ++i) a += qv[i] * kv[i];
    sc[idx] = a * (1.0f / 16.0f);
}

__global__ __launch_bounds__(256) void k_mha_softmax(float* __restrict__ sc) {
    int bh = blockIdx.x;           // B*4 rows of length TR
    float* row = sc + (long)bh * TR_;
    __shared__ float red[256];
    int tid = threadIdx.x;
    float mx = -1e30f;
    for (int i = tid; i < TR_; i += 256) mx = fmaxf(mx, row[i]);
    red[tid] = mx; __syncthreads();
    for (int st = 128; st > 0; st >>= 1) { if (tid < st) red[tid] = fmaxf(red[tid], red[tid + st]); __syncthreads(); }
    mx = red[0]; __syncthreads();
    float sum = 0.f;
    for (int i = tid; i < TR_; i += 256) sum += expf(row[i] - mx);
    red[tid] = sum; __syncthreads();
    for (int st = 128; st > 0; st >>= 1) { if (tid < st) red[tid] += red[tid + st]; __syncthreads(); }
    float inv = 1.0f / red[0];
    for (int i = tid; i < TR_; i += 256) row[i] = expf(row[i] - mx) * inv;
}

__global__ void k_mha_pool(const float* __restrict__ p, const float* __restrict__ vv,
                           float* __restrict__ o)
{
    int idx = blockIdx.x * 256 + threadIdx.x;  // B*4*256
    if (idx >= B_ * 4 * 256) return;
    int dd = idx % 256;
    int h  = (idx / 256) % 4;
    int b  = idx / (256 * 4);
    float a = 0.f;
    for (int l = 0; l < TR_; ++l)
        a += p[((long)(b * 4 + h)) * TR_ + l] * vv[((long)(b * TR_ + l)) * D_ + h * 256 + dd];
    o[(long)b * D_ + h * 256 + dd] = a;
}

__global__ void k_attn_residual(const float* __restrict__ x, const float* __restrict__ attn,
                                const float* __restrict__ bev_res, const float* __restrict__ r_res,
                                const float* __restrict__ mod,
                                const float* __restrict__ bgate, const float* __restrict__ rgate,
                                float* __restrict__ out)
{
    long idx = (long)blockIdx.x * 256 + threadIdx.x;
    if (idx >= (long)B_ * T_ * D_) return;
    int d = (int)(idx % D_);
    int b = (int)(idx / ((long)T_ * D_));
    float a = attn[idx] + tanhf(bgate[0]) * bev_res[(long)b * D_ + d]
                        + tanhf(rgate[0]) * r_res[(long)b * D_ + d];
    out[idx] = x[idx] + mod[(long)b * 6 * D_ + 2 * D_ + d] * a;
}

__global__ void k_ffn_residual(const float* __restrict__ x1, const float* __restrict__ h2,
                               const float* __restrict__ mod, float* __restrict__ out)
{
    long idx = (long)blockIdx.x * 256 + threadIdx.x;
    if (idx >= (long)B_ * T_ * D_) return;
    int d = (int)(idx % D_);
    int b = (int)(idx / ((long)T_ * D_));
    out[idx] = x1[idx] + mod[(long)b * 6 * D_ + 5 * D_ + d] * h2[idx];
}

// ---------------- host side ----------------

static void launch_gemm(hipStream_t st, int epi,
                        const __bf16* A, long sA, int lda,
                        const __bf16* W, long sW, int ldw,
                        const float* bias, const float* addSrc,
                        float* C, long sC, int ldc,
                        int M, int N, int K, int batch)
{
    dim3 g((M + BM_ - 1) / BM_, (N + BN_ - 1) / BN_, batch), b(256);
    switch (epi) {
        case 1: k_gemm_bf16<1><<<g, b, 0, st>>>(A, sA, lda, W, sW, ldw, bias, addSrc, C, sC, ldc, M, N, K); break;
        case 2: k_gemm_bf16<2><<<g, b, 0, st>>>(A, sA, lda, W, sW, ldw, bias, addSrc, C, sC, ldc, M, N, K); break;
        default: k_gemm_bf16<0><<<g, b, 0, st>>>(A, sA, lda, W, sW, ldw, bias, addSrc, C, sC, ldc, M, N, K); break;
    }
}

static void cvt(hipStream_t st, const float* in, __bf16* out, long n) {
    k_cvt_bf16<<<(unsigned)((n + 255) / 256), 256, 0, st>>>(in, out, n);
}

// input indices (setup_inputs leaf order)
enum {
    IX = 0, IBEV, IREASON, IACTION, ICOND,
    IQW, IQB, IADBW, IADBB, IADRW, IADRB, IADAW, IADAB, IADOW, IADOB,
    IKSW, IKSB, IVSW, IVSB, IKBW, IKBB, IVBW, IVBB,
    IKRW, IKRB, IVRW, IVRB, IKAW, IKAB, IVAW, IVAB,
    IOW, IOB, IBR1W, IBR1B, IBR2W, IBR2B, IRRPW, IRRPB,
    IF1W, IF1B, IF2W, IF2B, IADLW, IADLB,
    IQNW, IKNW, IGATE, IGATEA,
    ITS, ITBV, ITR, ITA, IBS, IBBV, IBR, IBA,
    IBRG, IRRG, IRQ, IMIW, IMIB, IMOW, IMOB,
    IFLNW, IFLNB, INPW, INPB
};

extern "C" void kernel_launch(void* const* d_in, const int* in_sizes, int n_in,
                              void* d_out, int out_size, void* d_ws, size_t ws_size,
                              hipStream_t stream)
{
    (void)in_sizes; (void)n_in; (void)out_size; (void)ws_size;
    hipStream_t st = stream;
    #define FIN(i) ((const float*)d_in[(i)])

    // ---- workspace bump allocator ----
    size_t off = 0;
    auto alloc = [&](size_t bytes) -> void* {
        void* p = (char*)d_ws + off;
        off = (off + bytes + 255) & ~(size_t)255;
        return p;
    };
    auto allocF = [&](long n) { return (float*)alloc((size_t)n * 4); };
    auto allocH = [&](long n) { return (__bf16*)alloc((size_t)n * 2); };

    const long MT = (long)B_ * T_;        // 2048 token rows

    // bf16 weight copies
    __bf16* wq    = allocH((long)D_ * D_);
    __bf16* wadb  = allocH((long)(D_ / 4) * D_);
    __bf16* wadr  = allocH((long)(D_ / 4) * D_);
    __bf16* wada  = allocH((long)(D_ / 4) * D_);
    __bf16* wado  = allocH((long)D_ * (D_ / 4));
    __bf16* wks   = allocH((long)D_ * D_);
    __bf16* wvs   = allocH((long)D_ * D_);
    __bf16* wkb   = allocH((long)D_ * D_);
    __bf16* wvb   = allocH((long)D_ * D_);
    __bf16* wkr   = allocH((long)D_ * D_);
    __bf16* wvr   = allocH((long)D_ * D_);
    __bf16* wka   = allocH((long)D_ * D_);
    __bf16* wva   = allocH((long)D_ * D_);
    __bf16* wo    = allocH((long)D_ * D_);
    __bf16* wf1   = allocH((long)FF_ * D_);
    __bf16* wf2   = allocH((long)D_ * FF_);
    __bf16* wmha  = allocH((long)3 * D_ * D_);

    // bf16 activations
    __bf16* xn_bf     = allocH(MT * D_);
    __bf16* bev_bf    = allocH((long)B_ * TB_ * D_);
    __bf16* reason_bf = allocH((long)B_ * TR_ * D_);
    __bf16* action_bf = allocH((long)B_ * TA_ * D_);
    __bf16* qad_bf    = allocH(MT * (D_ / 4));
    __bf16* qh        = allocH((long)BH_ * T_ * HD_);
    __bf16* kh        = allocH((long)BH_ * TB_ * HD_);   // max L
    __bf16* vt        = allocH((long)BH_ * HD_ * TB_);
    __bf16* probs     = allocH((long)BH_ * T_ * TB_);
    __bf16* merged_bf = allocH(MT * D_);
    __bf16* hln_bf    = allocH(MT * D_);
    __bf16* h1_bf     = allocH(MT * FF_);

    // f32 scratch
    float* cond_silu = allocF((long)B_ * D_);
    float* mod       = allocF((long)B_ * 6 * D_);
    float* qbase_f   = allocF(MT * D_);
    float* qad_f     = allocF(MT * (D_ / 4));
    float* qx_f      = allocF(MT * D_);
    float* kproj_f   = allocF((long)B_ * TB_ * D_);
    float* vproj_f   = allocF((long)B_ * TB_ * D_);
    float* scores    = allocF((long)BH_ * T_ * TB_);
    float* outs[4];
    for (int i = 0; i < 4; ++i) outs[i] = allocF((long)BH_ * T_ * HD_);
    float* merged    = allocF(MT * D_);
    float* attn_out  = allocF(MT * D_);
    float* bevpool   = allocF((long)B_ * D_);
    float* bevt1     = allocF((long)B_ * (D_ / 2));
    float* bev_res   = allocF((long)B_ * D_);
    float* qmha      = allocF(D_);
    float* kmha_f    = allocF((long)B_ * TR_ * D_);
    float* vmha_f    = allocF((long)B_ * TR_ * D_);
    float* mha_sc    = allocF((long)B_ * 4 * TR_);
    float* omha      = allocF((long)B_ * D_);
    float* rpool     = allocF((long)B_ * D_);
    float* r_res     = allocF((long)B_ * D_);
    float* x1        = allocF(MT * D_);
    float* h1_f      = allocF(MT * FF_);
    float* h2_f      = allocF(MT * D_);

    // ---- 1. AdaLN modulation: mod = silu(cond) @ adaln^T + b ----
    k_silu<<<(B_ * D_ + 255) / 256, 256, 0, st>>>(FIN(ICOND), cond_silu, B_ * D_);
    k_small_linear<0><<<(6 * D_ + 255) / 256, 256, 0, st>>>(
        cond_silu, FIN(IADLW), FIN(IADLB), mod, B_, 6 * D_, D_);

    // ---- 2. pre-LN + modulation -> xn (bf16) ----
    k_ln_mod<<<(unsigned)MT, 256, 0, st>>>(FIN(IX), FIN(INPW), FIN(INPB),
                                           mod, 0, D_, T_, xn_bf);

    // ---- 3. weight + token conversions ----
    cvt(st, FIN(IQW), wq, (long)D_ * D_);
    cvt(st, FIN(IADBW), wadb, (long)(D_ / 4) * D_);
    cvt(st, FIN(IADRW), wadr, (long)(D_ / 4) * D_);
    cvt(st, FIN(IADAW), wada, (long)(D_ / 4) * D_);
    cvt(st, FIN(IADOW), wado, (long)D_ * (D_ / 4));
    cvt(st, FIN(IKSW), wks, (long)D_ * D_);
    cvt(st, FIN(IVSW), wvs, (long)D_ * D_);
    cvt(st, FIN(IKBW), wkb, (long)D_ * D_);
    cvt(st, FIN(IVBW), wvb, (long)D_ * D_);
    cvt(st, FIN(IKRW), wkr, (long)D_ * D_);
    cvt(st, FIN(IVRW), wvr, (long)D_ * D_);
    cvt(st, FIN(IKAW), wka, (long)D_ * D_);
    cvt(st, FIN(IVAW), wva, (long)D_ * D_);
    cvt(st, FIN(IOW), wo, (long)D_ * D_);
    cvt(st, FIN(IF1W), wf1, (long)FF_ * D_);
    cvt(st, FIN(IF2W), wf2, (long)D_ * FF_);
    cvt(st, FIN(IMIW), wmha, (long)3 * D_ * D_);
    cvt(st, FIN(IBEV), bev_bf, (long)B_ * TB_ * D_);
    cvt(st, FIN(IREASON), reason_bf, (long)B_ * TR_ * D_);
    cvt(st, FIN(IACTION), action_bf, (long)B_ * TA_ * D_);

    // ---- 4. q_base projection ----
    launch_gemm(st, 0, xn_bf, 0, D_, wq, 0, D_, FIN(IQB), nullptr,
                qbase_f, 0, D_, (int)MT, D_, D_, 1);

    // ---- attention branch runner ----
    auto run_branch = [&](int bi, const float* q_src,
                          const __bf16* tok_bf, int Lsrc,
                          const __bf16* wk, const float* bk,
                          const __bf16* wv, const float* bv,
                          const float* tempP, const float* biasP)
    {
        // Q: rms+rope -> head-major bf16
        int qTokens = B_ * T_ * NH_;
        k_head_rms_rope<<<(qTokens + 7) / 8, 256, 0, st>>>(q_src, FIN(IQNW), qh, T_);
        // K projection + rms/rope
        launch_gemm(st, 0, tok_bf, 0, D_, wk, 0, D_, bk, nullptr,
                    kproj_f, 0, D_, B_ * Lsrc, D_, D_, 1);
        int kTokens = B_ * Lsrc * NH_;
        k_head_rms_rope<<<(kTokens + 7) / 8, 256, 0, st>>>(kproj_f, FIN(IKNW), kh, Lsrc);
        // V projection + transpose
        launch_gemm(st, 0, tok_bf, 0, D_, wv, 0, D_, bv, nullptr,
                    vproj_f, 0, D_, B_ * Lsrc, D_, D_, 1);
        long vtN = (long)B_ * NH_ * HD_ * Lsrc;
        k_vt_bf16<<<(unsigned)((vtN + 255) / 256), 256, 0, st>>>(vproj_f, vt, Lsrc);
        // scores = Qh @ Kh^T  (batched over BH heads)
        launch_gemm(st, 0, qh, (long)T_ * HD_, HD_, kh, (long)Lsrc * HD_, HD_,
                    nullptr, nullptr, scores, (long)T_ * Lsrc, Lsrc,
                    T_, Lsrc, HD_, BH_);
        // softmax rows
        k_softmax_row<<<(unsigned)((long)BH_ * T_), 256, 0, st>>>(scores, probs, Lsrc, tempP, biasP);
        // out = P @ V  (W operand = V^T, shape (HD, Lsrc))
        launch_gemm(st, 0, probs, (long)T_ * Lsrc, Lsrc, vt, (long)HD_ * Lsrc, Lsrc,
                    nullptr, nullptr, outs[bi], (long)T_ * HD_, HD_,
                    T_, HD_, Lsrc, BH_);
    };

    // ---- 5. self branch ----
    run_branch(0, qbase_f, xn_bf, T_, wks, FIN(IKSB), wvs, FIN(IVSB), FIN(ITS), FIN(IBS));

    // ---- 6-8. adapter branches (bev / reason / action) ----
    struct Ad { const __bf16* wad; const float* bad; const __bf16* tok; int L;
                const __bf16* wk; const float* bk; const __bf16* wv; const float* bv;
                const float* tp; const float* bp; };
    Ad ads[3] = {
        { wadb, FIN(IADBB), bev_bf,    TB_, wkb, FIN(IKBB), wvb, FIN(IVBB), FIN(ITBV), FIN(IBBV) },
        { wadr, FIN(IADRB), reason_bf, TR_, wkr, FIN(IKRB), wvr, FIN(IVRB), FIN(ITR),  FIN(IBR)  },
        { wada, FIN(IADAB), action_bf, TA_, wka, FIN(IKAB), wva, FIN(IVAB), FIN(ITA),  FIN(IBA)  },
    };
    for (int i = 0; i < 3; ++i) {
        // q_ad = tanh(xn @ Wad^T + b)
        launch_gemm(st, 1, xn_bf, 0, D_, ads[i].wad, 0, D_, ads[i].bad, nullptr,
                    qad_f, 0, D_ / 4, (int)MT, D_ / 4, D_, 1);
        cvt(st, qad_f, qad_bf, MT * (D_ / 4));
        // qX = q_base + q_ad @ Wout^T + b
        launch_gemm(st, 0, qad_bf, 0, D_ / 4, wado, 0, D_ / 4, FIN(IADOB), qbase_f,
                    qx_f, 0, D_, (int)MT, D_, D_ / 4, 1);
        run_branch(1 + i, qx_f, ads[i].tok, ads[i].L,
                   ads[i].wk, ads[i].bk, ads[i].wv, ads[i].bv, ads[i].tp, ads[i].bp);
    }

    // ---- 9. combine branches + output projection ----
    long nelem = MT * D_;
    k_combine<<<(unsigned)((nelem + 255) / 256), 256, 0, st>>>(
        outs[0], outs[1], outs[2], outs[3], FIN(IGATE), FIN(IGATEA), merged);
    cvt(st, merged, merged_bf, nelem);
    launch_gemm(st, 0, merged_bf, 0, D_, wo, 0, D_, FIN(IOB), nullptr,
                attn_out, 0, D_, (int)MT, D_, D_, 1);

    // ---- 10. BEV residual branch ----
    k_mean_pool<<<(B_ * D_ + 255) / 256, 256, 0, st>>>(FIN(IBEV), bevpool);
    k_small_linear<1><<<(D_ / 2 + 255) / 256, 256, 0, st>>>(
        bevpool, FIN(IBR1W), FIN(IBR1B), bevt1, B_, D_ / 2, D_);
    k_small_linear<0><<<(D_ + 255) / 256, 256, 0, st>>>(
        bevt1, FIN(IBR2W), FIN(IBR2B), bev_res, B_, D_, D_ / 2);

    // ---- 11. reasoning residual (pooled 4-head MHA) ----
    k_small_linear<0><<<(D_ + 255) / 256, 256, 0, st>>>(
        FIN(IRQ), FIN(IMIW), FIN(IMIB), qmha, 1, D_, D_);
    launch_gemm(st, 0, reason_bf, 0, D_, wmha + (long)D_ * D_, 0, D_,
                FIN(IMIB) + D_, nullptr, kmha_f, 0, D_, B_ * TR_, D_, D_, 1);
    launch_gemm(st, 0, reason_bf, 0, D_, wmha + (long)2 * D_ * D_, 0, D_,
                FIN(IMIB) + 2 * D_, nullptr, vmha_f, 0, D_, B_ * TR_, D_, D_, 1);
    k_mha_scores<<<(B_ * 4 * TR_ + 255) / 256, 256, 0, st>>>(qmha, kmha_f, mha_sc);
    k_mha_softmax<<<B_ * 4, 256, 0, st>>>(mha_sc);
    k_mha_pool<<<(B_ * 4 * 256 + 255) / 256, 256, 0, st>>>(mha_sc, vmha_f, omha);
    k_small_linear<0><<<(D_ + 255) / 256, 256, 0, st>>>(
        omha, FIN(IMOW), FIN(IMOB), rpool, B_, D_, D_);
    k_small_linear<0><<<(D_ + 255) / 256, 256, 0, st>>>(
        rpool, FIN(IRRPW), FIN(IRRPB), r_res, B_, D_, D_);

    // ---- 12. x1 = x + g_attn * (attn_out + gated residuals) ----
    k_attn_residual<<<(unsigned)((nelem + 255) / 256), 256, 0, st>>>(
        FIN(IX), attn_out, bev_res, r_res, mod, FIN(IBRG), FIN(IRRG), x1);

    // ---- 13. FFN ----
    k_ln_mod<<<(unsigned)MT, 256, 0, st>>>(x1, FIN(IFLNW), FIN(IFLNB),
                                           mod, 3 * D_, 4 * D_, T_, hln_bf);
    launch_gemm(st, 2, hln_bf, 0, D_, wf1, 0, D_, FIN(IF1B), nullptr,
                h1_f, 0, FF_, (int)MT, FF_, D_, 1);
    cvt(st, h1_f, h1_bf, MT * FF_);
    launch_gemm(st, 0, h1_bf, 0, FF_, wf2, 0, FF_, FIN(IF2B), nullptr,
                h2_f, 0, D_, (int)MT, D_, FF_, 1);

    // ---- 14. final residual -> d_out ----
    k_ffn_residual<<<(unsigned)((nelem + 255) / 256), 256, 0, st>>>(
        x1, h2_f, mod, (float*)d_out);

    #undef FIN
}
